// PrototypicalGeometricLoss_75806172774555
// MI455X (gfx1250) — matrix operand order repaired and checked by reading, not compiled, not run on hardware
//
#include <hip/hip_runtime.h>
#include <hip/hip_bf16.h>
#include <math.h>

// ---- problem constants (fixed by the reference) ----
#define BN 4096      // batch
#define SN 64        // sequence
#define DN 128       // feature dim
#define CN 10000     // classes
#define NT ((CN + 15) / 16)          // 625 N-tiles (exact: 10000 = 625*16)
#define MARGIN_F 2.0f

typedef _Float16 v16h __attribute__((ext_vector_type(16)));
typedef _Float16 v8h  __attribute__((ext_vector_type(8)));
typedef float    v8f  __attribute__((ext_vector_type(8)));

// ---------------------------------------------------------------------------
// zero fill
// ---------------------------------------------------------------------------
__global__ void zero_kernel(float* __restrict__ p, int n) {
    int i = blockIdx.x * blockDim.x + threadIdx.x;
    if (i < n) p[i] = 0.0f;
}

// ---------------------------------------------------------------------------
// K1: mean-pool over S, L2-normalize, emit f32 + f16 copies and g2 = |g|^2
//     one block per batch row, 128 threads (one per feature)
// ---------------------------------------------------------------------------
__global__ __launch_bounds__(128)
void pool_norm_kernel(const float* __restrict__ gs,
                      float* __restrict__ gn, _Float16* __restrict__ gn_h,
                      float* __restrict__ g2) {
    __shared__ float sd[DN];
    const int b = blockIdx.x;
    const int t = threadIdx.x;
    const float* src = gs + (size_t)b * SN * DN + t;
    float acc = 0.0f;
    #pragma unroll 8
    for (int s = 0; s < SN; ++s) acc += src[(size_t)s * DN];
    const float pooled = acc * (1.0f / (float)SN);
    sd[t] = pooled * pooled;
    __syncthreads();
    for (int s = DN / 2; s > 0; s >>= 1) {
        if (t < s) sd[t] += sd[t + s];
        __syncthreads();
    }
    const float ss = sd[0];
    const float denom = fmaxf(sqrtf(ss), 1e-12f);
    const float v = pooled / denom;
    gn[(size_t)b * DN + t]   = v;
    gn_h[(size_t)b * DN + t] = (_Float16)v;
    if (t == 0) g2[b] = ss / (denom * denom);
}

// ---------------------------------------------------------------------------
// K2: segment-sum scatter (counts + per-feature sums) via atomics
// ---------------------------------------------------------------------------
__global__ __launch_bounds__(128)
void scatter_kernel(const float* __restrict__ gn, const int* __restrict__ labels,
                    float* __restrict__ sums, float* __restrict__ counts) {
    const int b = blockIdx.x;
    const int t = threadIdx.x;
    const int lab = labels[b];
    atomicAdd(&sums[(size_t)lab * DN + t], gn[(size_t)b * DN + t]);
    if (t == 0) atomicAdd(&counts[lab], 1.0f);
}

// ---------------------------------------------------------------------------
// K3: EMA prototype update + renormalize; emit f32 + f16 copies and p2
//     one block per class, 128 threads
// ---------------------------------------------------------------------------
__global__ __launch_bounds__(128)
void proto_update_kernel(const float* __restrict__ protos,
                         const float* __restrict__ sums,
                         const float* __restrict__ counts,
                         float* __restrict__ pr, _Float16* __restrict__ pr_h,
                         float* __restrict__ p2v) {
    __shared__ float sd[DN];
    const int c = blockIdx.x;
    const int t = threadIdx.x;
    const float cnt = counts[c];
    const float po   = protos[(size_t)c * DN + t];
    const float mean = sums[(size_t)c * DN + t] / fmaxf(cnt, 1.0f);
    const float u = 0.9f * po + 0.1f * mean;
    sd[t] = u * u;
    __syncthreads();
    for (int s = DN / 2; s > 0; s >>= 1) {
        if (t < s) sd[t] += sd[t + s];
        __syncthreads();
    }
    const float denom = fmaxf(sqrtf(sd[0]), 1e-12f);
    const float p = (cnt > 0.0f) ? (u / denom) : po;
    pr[(size_t)c * DN + t]   = p;
    pr_h[(size_t)c * DN + t] = (_Float16)p;
    __syncthreads();               // protect sd before reuse
    sd[t] = p * p;
    __syncthreads();
    for (int s = DN / 2; s > 0; s >>= 1) {
        if (t < s) sd[t] += sd[t + s];
        __syncthreads();
    }
    if (t == 0) p2v[c] = sd[0];
}

// ---------------------------------------------------------------------------
// K4: exact f32 own-class distances -> intra sum + own-violation sum
//     one wave (32 lanes) per batch row
// ---------------------------------------------------------------------------
__global__ __launch_bounds__(256)
void own_dist_kernel(const float* __restrict__ gn, const float* __restrict__ pr,
                     const int* __restrict__ labels,
                     const float* __restrict__ g2, const float* __restrict__ p2,
                     float* __restrict__ scal) {
    const int wid  = (int)((blockIdx.x * blockDim.x + threadIdx.x) >> 5);
    const int lane = threadIdx.x & 31;
    if (wid >= BN) return;
    const int lab = labels[wid];
    const float4 x = ((const float4*)(gn + (size_t)wid * DN))[lane];
    const float4 y = ((const float4*)(pr + (size_t)lab * DN))[lane];
    float dot = x.x * y.x + x.y * y.y + x.z * y.z + x.w * y.w;
    #pragma unroll
    for (int off = 16; off > 0; off >>= 1) dot += __shfl_xor(dot, off, 32);
    if (lane == 0) {
        const float d2   = g2[wid] + p2[lab] - 2.0f * dot;
        const float dist = sqrtf(fmaxf(d2, 0.0f));
        atomicAdd(&scal[0], dist);                         // sum of own dists
        atomicAdd(&scal[1], fmaxf(MARGIN_F - dist, 0.0f)); // own violations
    }
}

// ---------------------------------------------------------------------------
// K5: fused distance-GEMM: gn_f16 @ protos_f16^T via v_wmma_f32_16x16x32_f16
//     with fused epilogue  part += relu(MARGIN - sqrt(max(g2+p2-2*dot,0))).
//     Block = 8 waves. Wave w keeps its 16x128 A-fragment in registers and
//     sweeps 4 N-tiles with INTERLEAVED accumulator chains: per K-step, 4
//     B-loads then 4 independent WMMAs, so WMMA->WMMA RAW latency on one
//     accumulator is hidden by the other three chains. Tail tiles (last
//     grid.y row only) clamp their B pointer to a valid tile and are
//     excluded from the epilogue sum, keeping the inner loop branch-free
//     (EXEC all-1s at every WMMA).
// ---------------------------------------------------------------------------
__global__ __launch_bounds__(256)
void gemm_viol_kernel(const _Float16* __restrict__ gn_h,
                      const _Float16* __restrict__ pr_h,
                      const float* __restrict__ g2,
                      const float* __restrict__ p2,
                      float* __restrict__ viol_accum) {
    __shared__ float bsum[8];
    const int w      = threadIdx.x >> 5;
    const int lane   = threadIdx.x & 31;
    const int mbase  = blockIdx.x * 16;
    const int ntile0 = (blockIdx.y * 8 + w) * 4;   // first of 4 N-tiles
    const int lrow   = lane & 15;
    const int hsel   = lane >> 4;

    // ---- A fragments (16x32 f16): per lane two 16B chunks,
    //      K=[0..7]/[8..15] (+16 for upper half-wave); kept live for reuse.
    const _Float16* arow = gn_h + (size_t)(mbase + lrow) * DN + hsel * 16;
    v16h a[4];
    #pragma unroll
    for (int j = 0; j < 4; ++j) {
        const v8h alo = *(const v8h*)(arow + 32 * j);
        const v8h ahi = *(const v8h*)(arow + 32 * j + 8);
        #pragma unroll
        for (int i = 0; i < 8; ++i) { a[j][i] = alo[i]; a[j][i + 8] = ahi[i]; }
    }
    // g2 for the 8 output rows this lane owns (same across all N-tiles)
    float g2r[8];
    #pragma unroll
    for (int i = 0; i < 8; ++i) g2r[i] = g2[mbase + i + 8 * hsel];

    // B row pointers per tile; clamp OOB tail tiles to a valid address.
    // B (32x16 f16, k order matching A): per lane one contiguous 16-half chunk of the
    // proto row; upper half-wave takes K+16.
    const _Float16* brow[4];
    #pragma unroll
    for (int t = 0; t < 4; ++t) {
        const int nt = (ntile0 + t < NT) ? (ntile0 + t) : (NT - 1);
        brow[t] = pr_h + (size_t)(nt * 16 + lrow) * DN + hsel * 16;
    }

    // ---- 4 interleaved accumulator chains over K=128
    v8f acc[4] = {};
    #pragma unroll
    for (int j = 0; j < 4; ++j) {
        v16h b0 = *(const v16h*)(brow[0] + 32 * j);
        v16h b1 = *(const v16h*)(brow[1] + 32 * j);
        v16h b2 = *(const v16h*)(brow[2] + 32 * j);
        v16h b3 = *(const v16h*)(brow[3] + 32 * j);
        acc[0] = __builtin_amdgcn_wmma_f32_16x16x32_f16(
            false, a[j], false, b0, (short)0, acc[0], false, false);
        acc[1] = __builtin_amdgcn_wmma_f32_16x16x32_f16(
            false, a[j], false, b1, (short)0, acc[1], false, false);
        acc[2] = __builtin_amdgcn_wmma_f32_16x16x32_f16(
            false, a[j], false, b2, (short)0, acc[2], false, false);
        acc[3] = __builtin_amdgcn_wmma_f32_16x16x32_f16(
            false, a[j], false, b3, (short)0, acc[3], false, false);
    }

    // ---- fused epilogue (skip clamped tail tiles)
    float part = 0.0f;
    #pragma unroll
    for (int t = 0; t < 4; ++t) {
        const int ntile = ntile0 + t;
        if (ntile < NT) {                  // wave-uniform
            const float p2c = p2[ntile * 16 + lrow];
            #pragma unroll
            for (int i = 0; i < 8; ++i) {
                const float d2   = g2r[i] + p2c - 2.0f * acc[t][i];
                // d2 in [0,4]: raw v_sqrt_f32, skip libm range fixup
                const float dist = __builtin_amdgcn_sqrtf(fmaxf(d2, 0.0f));
                part += fmaxf(MARGIN_F - dist, 0.0f);
            }
        }
    }
    // wave reduce, then one atomic per block
    #pragma unroll
    for (int off = 16; off > 0; off >>= 1) part += __shfl_xor(part, off, 32);
    if (lane == 0) bsum[w] = part;
    __syncthreads();
    if (threadIdx.x == 0) {
        float s = 0.0f;
        #pragma unroll
        for (int i = 0; i < 8; ++i) s += bsum[i];
        atomicAdd(viol_accum, s);
    }
}

// ---------------------------------------------------------------------------
// K6: finalize — std(ddof=1)/mean of simplex_volumes + combine 6 outputs
// ---------------------------------------------------------------------------
__global__ __launch_bounds__(256)
void finalize_kernel(const float* __restrict__ sv, const float* __restrict__ scal,
                     float* __restrict__ out) {
    __shared__ float s1[256], s2[256];
    const int t = threadIdx.x;
    float a = 0.0f, b = 0.0f;
    for (int i = t; i < BN; i += 256) { const float v = sv[i]; a += v; b += v * v; }
    s1[t] = a; s2[t] = b;
    __syncthreads();
    for (int s = 128; s > 0; s >>= 1) {
        if (t < s) { s1[t] += s1[t + s]; s2[t] += s2[t + s]; }
        __syncthreads();
    }
    if (t == 0) {
        const float Bf   = (float)BN;
        const float mean = s1[0] / Bf;
        const float var  = (s2[0] - Bf * mean * mean) / (Bf - 1.0f);
        const float stdv = sqrtf(fmaxf(var, 0.0f));
        const float intra = scal[0] / Bf;
        const float inter = (scal[2] - scal[1]) / (Bf * ((float)CN - 1.0f));
        const float vdl = -stdv;
        const float cr  = -mean;
        out[0] = 1.0f * intra + 2.0f * inter + 0.5f * vdl + 0.1f * cr;
        out[1] = intra;
        out[2] = inter;
        out[3] = vdl;
        out[4] = cr;
        out[5] = intra;   // mean_proto_dist == own_proto_dists.mean()
    }
}

// ---------------------------------------------------------------------------
extern "C" void kernel_launch(void* const* d_in, const int* in_sizes, int n_in,
                              void* d_out, int out_size, void* d_ws, size_t ws_size,
                              hipStream_t stream) {
    (void)in_sizes; (void)n_in; (void)out_size; (void)ws_size;
    const float* gs     = (const float*)d_in[0];  // [B,S,D]
    const float* sv     = (const float*)d_in[1];  // [B]
    const float* protos = (const float*)d_in[2];  // [C,D]
    const int*   labels = (const int*)d_in[3];    // [B]
    float* out = (float*)d_out;                   // 6 scalars

    // workspace layout (floats), halves placed 16B-aligned at the tail
    float* ws     = (float*)d_ws;
    float* gn     = ws;                                  // B*D
    float* pr     = gn + (size_t)BN * DN;                // C*D
    float* sums   = pr + (size_t)CN * DN;                // C*D
    float* counts = sums + (size_t)CN * DN;              // C
    float* g2     = counts + CN;                         // B
    float* p2     = g2 + BN;                             // C
    float* scal   = p2 + CN;                             // 16 (0:intra 1:own 2:viol)
    _Float16* gn_h = (_Float16*)(scal + 16);             // B*D halves
    _Float16* pr_h = gn_h + (size_t)BN * DN;             // C*D halves

    // zero the accumulated regions (sums+counts contiguous, then scalars)
    {
        const int n1 = CN * DN + CN;
        zero_kernel<<<(n1 + 255) / 256, 256, 0, stream>>>(sums, n1);
        zero_kernel<<<1, 256, 0, stream>>>(scal, 16);
    }

    pool_norm_kernel<<<BN, DN, 0, stream>>>(gs, gn, gn_h, g2);
    scatter_kernel<<<BN, DN, 0, stream>>>(gn, labels, sums, counts);
    proto_update_kernel<<<CN, DN, 0, stream>>>(protos, sums, counts, pr, pr_h, p2);
    own_dist_kernel<<<BN / 8, 256, 0, stream>>>(gn, pr, labels, g2, p2, scal);

    // 8 waves/block, 4 N-tiles/wave -> 32 N-tiles per block in y
    dim3 grid(BN / 16, (NT + 31) / 32);   // 256 x 20
    gemm_viol_kernel<<<grid, 256, 0, stream>>>(gn_h, pr_h, g2, p2, &scal[2]);

    finalize_kernel<<<1, 256, 0, stream>>>(sv, scal, out);
}
